// MultiheadAttention_60533269070099
// MI455X (gfx1250) — compile-verified, matrix operand order checked by reading, not run on hardware
//
#include <hip/hip_runtime.h>

typedef __attribute__((ext_vector_type(16))) __bf16 v16bf;
typedef __attribute__((ext_vector_type(8)))  float  v8f;

#define NQ_    2048
#define NKV_   2048
#define DMODEL 1024
#define NHEAD  16
#define HDIM   64

#if __has_builtin(__builtin_amdgcn_tensor_load_to_lds) && \
    __has_builtin(__builtin_amdgcn_s_wait_tensorcnt)
#define USE_TDM 1
typedef __attribute__((ext_vector_type(4))) unsigned tdm_v4u;
typedef __attribute__((ext_vector_type(8))) int      tdm_v8i;
typedef __attribute__((ext_vector_type(4))) int      tdm_v4i;
#else
#define USE_TDM 0
#endif

__device__ inline unsigned short f32_to_bf16(float f) {
  union { float f; unsigned u; } v; v.f = f;
  unsigned r = v.u + 0x7FFFu + ((v.u >> 16) & 1u);   // round-to-nearest-even
  return (unsigned short)(r >> 16);
}
__device__ inline unsigned pack2_bf16(float a, float b) {
  return (unsigned)f32_to_bf16(a) | ((unsigned)f32_to_bf16(b) << 16);
}

// Assemble a 16x32 bf16 fragment (A-layout; B uses same packing of B^T).
// Lane L holds row (L&15); VGPR p holds K pair at (p>>2)*16 + (p&3)*2 + (L>=16 ? 8:0).
__device__ inline v16bf load_frag16(const unsigned short* lds, int row, int stride,
                                    int kofs, int hi8) {
  union { v16bf v; unsigned u[8]; } f;
#pragma unroll
  for (int p = 0; p < 8; ++p) {
    int kb = ((p >> 2) * 16) + ((p & 3) * 2) + hi8;
    f.u[p] = *(const unsigned*)(lds + row * stride + kofs + kb);
  }
  return f.v;
}
__device__ inline v8f wmma_bf16(v16bf a, v16bf b, v8f c) {
  return __builtin_amdgcn_wmma_f32_16x16x32_bf16(false, a, false, b, (short)0, c,
                                                 false, false);
}

#if USE_TDM
// Issue a 2D TDM tile load (bf16 elements, data_size=2B) with LDS row padding.
// tile_w elements/row, tile_h rows; global row stride = stride_elems elements.
// pad_ic: pad interval code (2^(c+1) DWORDs); pad_ac: pad amount code (c+1 DWORDs).
__device__ inline void tdm_load_2d(const void* gptr, unsigned lds_addr,
                                   unsigned tile_w, unsigned tile_h,
                                   unsigned long long stride_elems,
                                   unsigned pad_ic, unsigned pad_ac) {
  unsigned long long ga = (unsigned long long)gptr;
  tdm_v4u g0;
  g0[0] = 1u;                                     // count=1 valid user descriptor
  g0[1] = lds_addr;                               // LDS byte address
  g0[2] = (unsigned)(ga & 0xFFFFFFFFull);         // global_addr[31:0]
  g0[3] = (unsigned)((ga >> 32) & 0x1FFFFFFull)   // global_addr[56:32]
        | (2u << 30);                             // type = 2 ("image")
  tdm_v8i g1;
  g1[0] = (int)((1u << 16)                        // data_size = 2 bytes
              | (1u << 20)                        // pad_enable
              | (pad_ic << 22) | (pad_ac << 25));
  g1[1] = (int)((tile_w & 0xFFFFu) << 16);        // tensor_dim0[15:0] (== tile_w)
  g1[2] = (int)(((tile_w >> 16) & 0xFFFFu) | ((tile_h & 0xFFFFu) << 16)); // dim0 hi | dim1 lo
  g1[3] = (int)(((tile_h >> 16) & 0xFFFFu) | ((tile_w & 0xFFFFu) << 16)); // dim1 hi | tile_dim0
  g1[4] = (int)(tile_h & 0xFFFFu);                // tile_dim1 (tile_dim2 = 0)
  g1[5] = (int)(unsigned)(stride_elems & 0xFFFFFFFFull); // tensor_dim0_stride lo
  g1[6] = (int)(unsigned)((stride_elems >> 32) & 0xFFFFull);
  g1[7] = 0;                                      // tensor_dim1_stride = 0 (2D)
  tdm_v4i gz4 = {0, 0, 0, 0};
  tdm_v8i gz8 = {0, 0, 0, 0, 0, 0, 0, 0};
  // clang-23 / amdgpu-toolchain 6-arg form:
  // (v4u g0, v8i g1, v4i g2, v4i g3, v8i g4, i32 cpol)
  __builtin_amdgcn_tensor_load_to_lds(g0, g1, gz4, gz4, gz8, 0);
}
__device__ inline unsigned lds_addr_of(const void* p) {
  return (unsigned)(unsigned long long)p;          // low 32 bits = LDS offset
}
#endif

// ---------------------------------------------------------------------------
// GEMM: Y = X @ W^T + bias.  X: [8192,1024] (f32, or bf16 when mode==3),
// W: [1024,1024] f32 row-major (so B(k,n)=W[n,k] -> contiguous K reads).
// mode 0: Y->qh  bf16   mode 1: Y->kh  bf16   mode 2: Y->vhT bf16
// mode 3: Y->out [8192,1024] f32
// ---------------------------------------------------------------------------
__global__ __launch_bounds__(256)
void mha_gemm_kernel(const float* __restrict__ Xf, const unsigned short* __restrict__ Xb,
                     const float* __restrict__ W, const float* __restrict__ bias,
                     unsigned short* __restrict__ Yb, float* __restrict__ Yf, int mode)
{
  __shared__ unsigned short Xs[128 * 40];
  __shared__ unsigned short Ws[128 * 40];

  const int t = threadIdx.x;
  const int lane = t & 31, wid = t >> 5;
  const int hi8 = (lane >> 4) ? 8 : 0;
  const int wm = wid >> 1, wn = wid & 1;
  const int m0 = blockIdx.y * 128, n0 = blockIdx.x * 128;

  v8f zero = {};
  v8f acc[2][4];
#pragma unroll
  for (int i = 0; i < 2; ++i)
#pragma unroll
    for (int j = 0; j < 4; ++j) acc[i][j] = zero;

  for (int k0 = 0; k0 < DMODEL; k0 += 32) {
    __syncthreads();
#pragma unroll
    for (int p = 0; p < 4; ++p) {
      int row = p * 32 + (t >> 3);
      int c4  = (t & 7) * 4;
      uint2 xw;
      if (mode != 3) {
        float4 xv = *(const float4*)(Xf + (size_t)(m0 + row) * DMODEL + k0 + c4);
        xw.x = pack2_bf16(xv.x, xv.y); xw.y = pack2_bf16(xv.z, xv.w);
      } else {
        xw = *(const uint2*)(Xb + (size_t)(m0 + row) * DMODEL + k0 + c4);
      }
      *(uint2*)&Xs[row * 40 + c4] = xw;

      float4 wv = *(const float4*)(W + (size_t)(n0 + row) * DMODEL + k0 + c4);
      uint2 ww; ww.x = pack2_bf16(wv.x, wv.y); ww.y = pack2_bf16(wv.z, wv.w);
      *(uint2*)&Ws[row * 40 + c4] = ww;
    }
    if (k0 + 32 < DMODEL) {  // hint next W tile toward the caches
      __builtin_prefetch(W + (size_t)(n0 + (t >> 3)) * DMODEL + k0 + 32, 0, 3);
    }
    __syncthreads();

    v16bf a0 = load_frag16(Xs, wm * 32 +      (lane & 15), 40, 0, hi8);
    v16bf a1 = load_frag16(Xs, wm * 32 + 16 + (lane & 15), 40, 0, hi8);
#pragma unroll
    for (int j = 0; j < 4; ++j) {
      v16bf b = load_frag16(Ws, wn * 64 + j * 16 + (lane & 15), 40, 0, hi8);
      acc[0][j] = wmma_bf16(a0, b, acc[0][j]);
      acc[1][j] = wmma_bf16(a1, b, acc[1][j]);
    }
  }

  // Epilogue: C-frag element (m = r + hi8, n = lane&15)
#pragma unroll
  for (int i = 0; i < 2; ++i) {
#pragma unroll
    for (int j = 0; j < 4; ++j) {
      int gn = n0 + wn * 64 + j * 16 + (lane & 15);
      float bv = bias[gn];
#pragma unroll
      for (int r = 0; r < 8; ++r) {
        int gm = m0 + wm * 32 + i * 16 + r + hi8;
        float val = acc[i][j][r] + bv;
        if (mode == 3) {
          Yf[(size_t)gm * DMODEL + gn] = val;
        } else {
          int batch = gm >> 11, nq = gm & 2047;
          int h = gn >> 6, hd = gn & 63;
          if (mode == 2)
            Yb[(((size_t)batch * NHEAD + h) * HDIM + hd) * NKV_ + nq] = f32_to_bf16(val);
          else
            Yb[(((size_t)batch * NHEAD + h) * NQ_ + nq) * HDIM + hd]  = f32_to_bf16(val);
        }
      }
    }
  }
}

// ---------------------------------------------------------------------------
// Flash attention: block = one (b,h), 128 q-rows, 8 waves (16 q-rows each).
// K/V^T chunks staged by the Tensor Data Mover with double buffering:
// wave 0 issues chunk it+1, s_wait_tensorcnt(2) retires chunk it, barrier,
// then 4 QK^T WMMAs -> online softmax -> P via LDS -> 4 PV WMMAs.
// ---------------------------------------------------------------------------
__global__ __launch_bounds__(256)
void mha_attention_kernel(const unsigned short* __restrict__ qh,
                          const unsigned short* __restrict__ kh,
                          const unsigned short* __restrict__ vhT,
                          unsigned short* __restrict__ attn)
{
  __shared__ unsigned short Qs[128 * 72];
  __shared__ unsigned short Ks[2][32 * 72];
  __shared__ unsigned short Vts[2][64 * 40];
  __shared__ unsigned short Ps[8 * 16 * 40];

  const int t = threadIdx.x, lane = t & 31, wid = t >> 5;
  const int hi8 = (lane >> 4) ? 8 : 0;
  const int q0 = blockIdx.x * 128;
  const int bh = blockIdx.y;
  const size_t qbase = (size_t)bh * NQ_ * HDIM;
  const size_t kbase = (size_t)bh * NKV_ * HDIM;
  const size_t vbase = (size_t)bh * HDIM * NKV_;

  // ---- Stage the 128x64 Q tile (row stride 64 -> padded LDS stride 72) ----
#if USE_TDM
  if (wid == 0) {
    tdm_load_2d(qh + qbase + (size_t)q0 * HDIM, lds_addr_of(&Qs[0]),
                /*tile_w=*/64, /*tile_h=*/128, /*stride=*/64,
                /*pad_ic: 32 DW=*/4, /*pad_ac: 4 DW=*/3);
    __builtin_amdgcn_s_wait_tensorcnt(0);
  }
#else
#pragma unroll
  for (int p = 0; p < 8; ++p) {
    int lin = p * 256 + t;
    int row = lin >> 4, c4 = (lin & 15) * 4;
    *(uint2*)&Qs[row * 72 + c4] =
        *(const uint2*)(qh + qbase + (size_t)(q0 + row) * HDIM + c4);
  }
#endif
  __syncthreads();
  v16bf qa0 = load_frag16(Qs, wid * 16 + (lane & 15), 72, 0,  hi8);
  v16bf qa1 = load_frag16(Qs, wid * 16 + (lane & 15), 72, 32, hi8);

  v8f zero = {};
  v8f acc[4];
#pragma unroll
  for (int j = 0; j < 4; ++j) acc[j] = zero;
  float mrow[8], lrow[8];
#pragma unroll
  for (int r = 0; r < 8; ++r) { mrow[r] = -1e30f; lrow[r] = 0.f; }

  const int pbase = wid * 16 * 40;
  const int NIT = NKV_ / 32;
  int buf = 0;

#if USE_TDM
  if (wid == 0) {  // prologue: chunk 0 into buffer 0
    tdm_load_2d(kh + kbase, lds_addr_of(&Ks[0][0]), 64, 32, 64, 4, 3);
    tdm_load_2d(vhT + vbase, lds_addr_of(&Vts[0][0]), 32, 64, NKV_, 3, 3);
  }
#endif

  for (int it = 0; it < NIT; ++it) {
    const int k0 = it * 32;
    __syncthreads();  // all reads of buf^1 from the previous iteration are done
#if USE_TDM
    if (wid == 0) {
      if (it + 1 < NIT) {
        const int kn = (it + 1) * 32;
        tdm_load_2d(kh + kbase + (size_t)kn * HDIM, lds_addr_of(&Ks[buf ^ 1][0]),
                    64, 32, 64, 4, 3);
        tdm_load_2d(vhT + vbase + kn, lds_addr_of(&Vts[buf ^ 1][0]),
                    32, 64, NKV_, 3, 3);
        __builtin_amdgcn_s_wait_tensorcnt(2);  // retire chunk `it`, keep it+1 inflight
      } else {
        __builtin_amdgcn_s_wait_tensorcnt(0);
      }
    }
#else
#pragma unroll
    for (int p = 0; p < 2; ++p) {
      int lin = p * 256 + t;
      int row = lin >> 4, c4 = (lin & 15) * 4;
      *(uint2*)&Ks[buf][row * 72 + c4] =
          *(const uint2*)(kh + kbase + (size_t)(k0 + row) * HDIM + c4);
    }
#pragma unroll
    for (int p = 0; p < 2; ++p) {
      int lin = p * 256 + t;
      int d = lin >> 3, c4 = (lin & 7) * 4;
      *(uint2*)&Vts[buf][d * 40 + c4] =
          *(const uint2*)(vhT + vbase + (size_t)d * NKV_ + k0 + c4);
    }
#endif
    __syncthreads();  // chunk `it` resident in Ks[buf]/Vts[buf]

    // S[16x32] = Q[16x64] @ K^T : two 16-col subtiles, K-depth 64 = 2 WMMAs each
    v8f s0 = zero, s1 = zero;
    {
      v16bf kb00 = load_frag16(Ks[buf],      (lane & 15), 72, 0,  hi8);
      v16bf kb01 = load_frag16(Ks[buf],      (lane & 15), 72, 32, hi8);
      s0 = wmma_bf16(qa0, kb00, s0);
      s0 = wmma_bf16(qa1, kb01, s0);
      v16bf kb10 = load_frag16(Ks[buf], 16 + (lane & 15), 72, 0,  hi8);
      v16bf kb11 = load_frag16(Ks[buf], 16 + (lane & 15), 72, 32, hi8);
      s1 = wmma_bf16(qa0, kb10, s1);
      s1 = wmma_bf16(qa1, kb11, s1);
    }

    // Online softmax; C-frag row m = r + hi8 lives across the 16 lanes of a half
#pragma unroll
    for (int r = 0; r < 8; ++r) {
      float a = s0[r] * 0.125f;      // 1/sqrt(64)
      float b = s1[r] * 0.125f;
      float rm = fmaxf(a, b);
#pragma unroll
      for (int msk = 1; msk < 16; msk <<= 1)
        rm = fmaxf(rm, __shfl_xor(rm, msk, 32));
      float nm   = fmaxf(mrow[r], rm);
      float corr = __expf(mrow[r] - nm);
      mrow[r] = nm;
      float p0 = __expf(a - nm), p1 = __expf(b - nm);
      float rs = p0 + p1;
#pragma unroll
      for (int msk = 1; msk < 16; msk <<= 1)
        rs += __shfl_xor(rs, msk, 32);
      lrow[r] = lrow[r] * corr + rs;
#pragma unroll
      for (int j = 0; j < 4; ++j) acc[j][r] *= corr;
      int m = r + hi8, n = lane & 15;
      Ps[pbase + m * 40 + n]      = f32_to_bf16(p0);
      Ps[pbase + m * 40 + 16 + n] = f32_to_bf16(p1);
    }
    __syncthreads();

    // O[16x64] += P[16x32] @ V[32x64]  (V^T rows are contiguous in K)
    v16bf pf = load_frag16(Ps + pbase, (lane & 15), 40, 0, hi8);
#pragma unroll
    for (int j = 0; j < 4; ++j) {
      v16bf vb = load_frag16(Vts[buf], j * 16 + (lane & 15), 40, 0, hi8);
      acc[j] = wmma_bf16(pf, vb, acc[j]);
    }
#if USE_TDM
    buf ^= 1;
#endif
  }

  // Normalize and write bf16 output in [B, NQ, D] layout for the O-projection
  const int batch = bh >> 4, h = bh & 15;
#pragma unroll
  for (int r = 0; r < 8; ++r) {
    float inv = 1.0f / lrow[r];
    int qrow = q0 + wid * 16 + r + hi8;
#pragma unroll
    for (int j = 0; j < 4; ++j) {
      int col = h * HDIM + j * 16 + (lane & 15);
      attn[((size_t)batch * NQ_ + qrow) * DMODEL + col] = f32_to_bf16(acc[j][r] * inv);
    }
  }
}

extern "C" void kernel_launch(void* const* d_in, const int* in_sizes, int n_in,
                              void* d_out, int out_size, void* d_ws, size_t ws_size,
                              hipStream_t stream) {
  const float* q  = (const float*)d_in[0];
  const float* k  = (const float*)d_in[1];
  const float* v  = (const float*)d_in[2];
  const float* Wq = (const float*)d_in[3];
  const float* bq = (const float*)d_in[4];
  const float* Wk = (const float*)d_in[5];
  const float* bk = (const float*)d_in[6];
  const float* Wv = (const float*)d_in[7];
  const float* bv = (const float*)d_in[8];
  const float* Wo = (const float*)d_in[9];
  const float* bo = (const float*)d_in[10];

  const size_t HT = (size_t)4 * NHEAD * NQ_ * HDIM;  // 8,388,608 bf16 elements
  unsigned short* qh   = (unsigned short*)d_ws;
  unsigned short* kh   = qh  + HT;
  unsigned short* vhT  = kh  + HT;
  unsigned short* attn = vhT + HT;

  dim3 blk(256);
  dim3 gproj(DMODEL / 128, (4 * NQ_) / 128);   // (8, 64)

  mha_gemm_kernel<<<gproj, blk, 0, stream>>>(q, nullptr, Wq, bq, qh,  nullptr, 0);
  mha_gemm_kernel<<<gproj, blk, 0, stream>>>(k, nullptr, Wk, bk, kh,  nullptr, 1);
  mha_gemm_kernel<<<gproj, blk, 0, stream>>>(v, nullptr, Wv, bv, vhT, nullptr, 2);

  mha_attention_kernel<<<dim3(NQ_ / 128, 4 * NHEAD), blk, 0, stream>>>(qh, kh, vhT, attn);

  mha_gemm_kernel<<<gproj, blk, 0, stream>>>(nullptr, attn, Wo, bo, nullptr,
                                             (float*)d_out, 3);
}
